// GCN_29678224015760
// MI455X (gfx1250) — matrix operand order repaired and checked by reading, not compiled, not run on hardware
//
#include <hip/hip_runtime.h>
#include <hip/hip_bf16.h>
#include <stdint.h>

// Problem constants (match reference)
#define N_NODES  50000
#define N_EDGES  600000
#define N_FEAT   64
#define DHID     128
#define N_GRAPHS 1024

typedef __attribute__((ext_vector_type(16))) __bf16 v16bf;
typedef __attribute__((ext_vector_type(8)))  float  v8f;

union Frag { v16bf v; uint4 u[2]; };

// ---------------------------------------------------------------------------
// Scatter-add: agg[dst[e], :] += x[src[e], :]  (float4 per thread, L2 atomics)
// ---------------------------------------------------------------------------
__global__ void __launch_bounds__(256)
scatter_add_kernel(const float* __restrict__ x,
                   const int*   __restrict__ src,
                   const int*   __restrict__ dst,
                   float*       __restrict__ agg,
                   int F) {
    const int chunks = F >> 2;  // float4 chunks per row
    long long i = (long long)blockIdx.x * blockDim.x + threadIdx.x;
    const long long total = (long long)N_EDGES * chunks;
    if (i >= total) return;
    const int e = (int)(i / chunks);
    const int c = (int)(i % chunks) << 2;
    const int s = src[e];
    const int d = dst[e];
    const float4 v = *(const float4*)(x + (size_t)s * F + c);
    float* o = agg + (size_t)d * F + c;
    atomicAdd(o + 0, v.x);
    atomicAdd(o + 1, v.y);
    atomicAdd(o + 2, v.z);
    atomicAdd(o + 3, v.w);
}

// ---------------------------------------------------------------------------
// Fused (x + agg) -> bf16 activation matrix
// ---------------------------------------------------------------------------
__global__ void __launch_bounds__(256)
add_cvt_kernel(const float* __restrict__ x,
               const float* __restrict__ agg,
               __bf16*      __restrict__ abf,
               long long n) {
    long long i = (long long)blockIdx.x * blockDim.x + threadIdx.x;
    if (i >= n) return;
    abf[i] = (__bf16)(x[i] + agg[i]);
}

// ---------------------------------------------------------------------------
// Weight transpose + convert: wt[n*K + k] = bf16(w[k*128 + n])   (N = 128)
// ---------------------------------------------------------------------------
__global__ void __launch_bounds__(256)
wt_cvt_kernel(const float* __restrict__ w,
              __bf16*      __restrict__ wt,
              int K) {
    int i = blockIdx.x * blockDim.x + threadIdx.x;
    if (i >= K * DHID) return;
    const int n = i / K;
    const int k = i % K;
    wt[(size_t)n * K + k] = (__bf16)w[(size_t)k * DHID + n];
}

// ---------------------------------------------------------------------------
// WMMA GEMM + bias + ReLU:  C[M x 128] = relu(A[M x K] * Bt[128 x K]^T + b)
// A, Bt in bf16; C fp32.  One wave per 16x16 tile; 8 waves/block = 8 N-tiles;
// grid.x = M/16 (M is a multiple of 16).  EXEC all-1s: no divergence.
//
// A per-lane layout (ISA 7.12.2, 16-bit A 16x32):
//   m = lane&15 ; lanes>=16 shift K by 8 in each half:
//   elems 0..7  -> K = kb + 8*(lane>>4) + 0..7        (one 16B load)
//   elems 8..15 -> K = kb + 16 + 8*(lane>>4) + 0..7   (one 16B load)
// B per-lane layout (same K striping as A):
//   n = lane&15 ; elems 0..7 -> K = kb + 8*(lane>>4) + 0..7, elems 8..15 -> K = kb + 16 + 8*(lane>>4) + 0..7
// ---------------------------------------------------------------------------
__global__ void __launch_bounds__(256)
gemm_bias_relu_kernel(const __bf16* __restrict__ A,    // M x K, row-major
                      const __bf16* __restrict__ Bt,   // 128 x K, N-major
                      const float*  __restrict__ bias, // 128
                      float*        __restrict__ C,    // M x 128
                      int K) {
    const int lane  = threadIdx.x & 31;
    const int nTile = threadIdx.x >> 5;   // 0..7
    const int mTile = blockIdx.x;

    const int mRow = mTile * 16 + (lane & 15);
    const int nCol = nTile * 16 + (lane & 15);
    const int aShift = (lane >> 4) << 3;  // 0 or 8
    const int bShift = (lane >> 4) << 3;  // 0 or 8

    const __bf16* aBase = A  + (size_t)mRow * K + aShift;
    const __bf16* bBase = Bt + (size_t)nCol * K + bShift;

    v8f acc = {};
    for (int kb = 0; kb < K; kb += 32) {
        Frag a, b;
        a.u[0] = *(const uint4*)(aBase + kb);
        a.u[1] = *(const uint4*)(aBase + kb + 16);
        b.u[0] = *(const uint4*)(bBase + kb);
        b.u[1] = *(const uint4*)(bBase + kb + 16);
        if (kb + 32 < K) {
            __builtin_prefetch(aBase + kb + 32, 0, 3);   // global_prefetch_b8
            __builtin_prefetch(bBase + kb + 32, 0, 3);
        }
        // (neg_a, A, neg_b, B, c_mod, C, reuse_a, reuse_b)
        acc = __builtin_amdgcn_wmma_f32_16x16x32_bf16(
            false, a.v, false, b.v, (short)0, acc, false, false);
    }

    // D layout: VGPR v, lane L -> M = v + 8*(L>=16), N = L&15
    const int mBase = mTile * 16 + ((lane >> 4) << 3);
    const int n     = nTile * 16 + (lane & 15);
    const float bv  = bias[n];
#pragma unroll
    for (int v = 0; v < 8; ++v) {
        const float r = acc[v] + bv;
        C[(size_t)(mBase + v) * DHID + n] = fmaxf(r, 0.0f);
    }
}

// ---------------------------------------------------------------------------
// Global mean pool: atomic accumulate per-graph sums and counts
// ---------------------------------------------------------------------------
__global__ void __launch_bounds__(256)
pool_accum_kernel(const float* __restrict__ h,
                  const int*   __restrict__ batch,
                  float*       __restrict__ gsum,
                  float*       __restrict__ gcnt) {
    long long i = (long long)blockIdx.x * blockDim.x + threadIdx.x;
    const long long total = (long long)N_NODES * DHID;
    if (i >= total) return;
    const int node = (int)(i >> 7);      // DHID == 128
    const int f    = (int)(i & 127);
    const int g    = batch[node];
    atomicAdd(&gsum[(size_t)g * DHID + f], h[i]);
    if (f == 0) atomicAdd(&gcnt[g], 1.0f);
}

__global__ void __launch_bounds__(256)
pool_finalize_kernel(const float* __restrict__ gsum,
                     const float* __restrict__ gcnt,
                     __bf16*      __restrict__ pbf) {
    int i = blockIdx.x * blockDim.x + threadIdx.x;
    if (i >= N_GRAPHS * DHID) return;
    const float c = fmaxf(gcnt[i >> 7], 1.0f);
    pbf[i] = (__bf16)(gsum[i] / c);
}

// ---------------------------------------------------------------------------
// Host launch
// ---------------------------------------------------------------------------
static inline size_t align256(size_t v) { return (v + 255) & ~(size_t)255; }

extern "C" void kernel_launch(void* const* d_in, const int* in_sizes, int n_in,
                              void* d_out, int out_size, void* d_ws, size_t ws_size,
                              hipStream_t stream) {
    (void)in_sizes; (void)n_in; (void)out_size; (void)ws_size;

    const float* x     = (const float*)d_in[0];
    const int*   ei    = (const int*)  d_in[1];
    const int*   batch = (const int*)  d_in[2];
    const float* W[4]  = { (const float*)d_in[3], (const float*)d_in[5],
                           (const float*)d_in[7], (const float*)d_in[9] };
    const float* B[4]  = { (const float*)d_in[4], (const float*)d_in[6],
                           (const float*)d_in[8], (const float*)d_in[10] };
    const float* wp    = (const float*)d_in[11];
    const float* bp    = (const float*)d_in[12];
    float*       out   = (float*)d_out;

    const int* src = ei;
    const int* dst = ei + N_EDGES;

    // Workspace carve
    char* ws = (char*)d_ws;
    size_t off = 0;
    float*  agg  = (float*) (ws + off); off = align256(off + (size_t)N_NODES * DHID * sizeof(float));
    float*  h    = (float*) (ws + off); off = align256(off + (size_t)N_NODES * DHID * sizeof(float));
    __bf16* abf  = (__bf16*)(ws + off); off = align256(off + (size_t)N_NODES * DHID * sizeof(__bf16));
    __bf16* wtbf = (__bf16*)(ws + off); off = align256(off + (size_t)DHID * DHID * sizeof(__bf16));
    float*  gsum = (float*) (ws + off); off = align256(off + (size_t)N_GRAPHS * DHID * sizeof(float));
    float*  gcnt = (float*) (ws + off); off = align256(off + (size_t)N_GRAPHS * sizeof(float));
    __bf16* pbf  = (__bf16*)(ws + off); off = align256(off + (size_t)N_GRAPHS * DHID * sizeof(__bf16));

    const dim3 blk(256);
    const int mTiles = N_NODES / 16;  // 50000 / 16 = 3125 exactly

    // ---------------- Layer 0 (K = 64) ----------------
    {
        const int K = N_FEAT;
        hipMemsetAsync(agg, 0, (size_t)N_NODES * K * sizeof(float), stream);
        long long sc_total = (long long)N_EDGES * (K >> 2);
        scatter_add_kernel<<<(unsigned)((sc_total + 255) / 256), blk, 0, stream>>>(x, src, dst, agg, K);
        long long ac_total = (long long)N_NODES * K;
        add_cvt_kernel<<<(unsigned)((ac_total + 255) / 256), blk, 0, stream>>>(x, agg, abf, ac_total);
        wt_cvt_kernel<<<(K * DHID + 255) / 256, blk, 0, stream>>>(W[0], wtbf, K);
        gemm_bias_relu_kernel<<<mTiles, blk, 0, stream>>>(abf, wtbf, B[0], h, K);
    }

    // ---------------- Layers 1..3 (K = 128) ----------------
    for (int l = 1; l < 4; ++l) {
        const int K = DHID;
        hipMemsetAsync(agg, 0, (size_t)N_NODES * K * sizeof(float), stream);
        long long sc_total = (long long)N_EDGES * (K >> 2);
        scatter_add_kernel<<<(unsigned)((sc_total + 255) / 256), blk, 0, stream>>>(h, src, dst, agg, K);
        long long ac_total = (long long)N_NODES * K;
        add_cvt_kernel<<<(unsigned)((ac_total + 255) / 256), blk, 0, stream>>>(h, agg, abf, ac_total);
        wt_cvt_kernel<<<(K * DHID + 255) / 256, blk, 0, stream>>>(W[l], wtbf, K);
        gemm_bias_relu_kernel<<<mTiles, blk, 0, stream>>>(abf, wtbf, B[l], h, K);
    }

    // ---------------- Global mean pool + post-MLP ----------------
    hipMemsetAsync(gsum, 0, (size_t)N_GRAPHS * DHID * sizeof(float), stream);
    hipMemsetAsync(gcnt, 0, (size_t)N_GRAPHS * sizeof(float), stream);
    long long pa_total = (long long)N_NODES * DHID;
    pool_accum_kernel<<<(unsigned)((pa_total + 255) / 256), blk, 0, stream>>>(h, batch, gsum, gcnt);
    pool_finalize_kernel<<<(N_GRAPHS * DHID + 255) / 256, blk, 0, stream>>>(gsum, gcnt, pbf);
    wt_cvt_kernel<<<(DHID * DHID + 255) / 256, blk, 0, stream>>>(wp, wtbf, DHID);
    gemm_bias_relu_kernel<<<N_GRAPHS / 16, blk, 0, stream>>>(pbf, wtbf, bp, out, DHID);
}
